// MultiheadAttention_54760833024666
// MI455X (gfx1250) — compile-verified, matrix operand order checked by reading, not run on hardware
//
#include <hip/hip_runtime.h>
#include <hip/hip_bf16.h>

// ---------------------------------------------------------------------------
// MHA block for MI455X (gfx1250, wave32, WMMA).
// All GEMM-shaped math goes through v_wmma_f32_16x16x32_bf16.
// Pipeline: cvt(f32->bf16) -> QKV GEMM(+clip) -> q/k LayerNorm ->
//           flash attention -> out GEMM (fp32 out).
// GEMM: 128x256 block tile, 64x64 wave tile, K-step 64 ->
//       32 WMMAs per 16 LDS fragment loads per wave per K-step.
// Workspace layout (bf16): xb[M*D] wqkvb[3D*D] woutb[D*D] qkvb[M*3D] ctxb[M*D]
// ---------------------------------------------------------------------------

typedef __bf16 bf16_t;
typedef __attribute__((ext_vector_type(16))) __bf16 v16bf;
typedef __attribute__((ext_vector_type(8)))  __bf16 v8bf;
typedef __attribute__((ext_vector_type(8)))  float  v8f;

constexpr int BB  = 2;
constexpr int SS  = 2048;
constexpr int DD  = 2048;
constexpr int HH  = 16;
constexpr int DHH = 128;           // head dim
constexpr int MM  = BB * SS;       // 4096 token rows
constexpr float CLIPV = 8.0f;
constexpr float NEGBIG = -3.0e38f;

// ---------------------------------------------------------------------------
// WMMA helpers
// ---------------------------------------------------------------------------
__device__ inline v8f wmma_bf16(v16bf a, v16bf b, v8f c) {
  // D(16x16,f32) = A(16x32,bf16) * B(32x16,bf16) + C
  return __builtin_amdgcn_wmma_f32_16x16x32_bf16(
      false, a, false, b, (short)0, c, false, false);
}

// Load a 16x32 A/B fragment from a row-major [rows x ld] bf16 tile.
// Lane l: row = l&15 ; K-half = (l>>4)*8 ; elems 0..7 -> K = half+0..7,
// elems 8..15 -> K = 16+half+0..7  (CDNA5 ISA 16-bit A layout; B mirrors it
// as [N,K], so the same loader serves W^T GEMMs).
__device__ inline v16bf frag_ld(const bf16_t* __restrict__ p, int row0, int ld,
                                int koff) {
  const int lane = threadIdx.x & 31;
  const int r    = lane & 15;
  const int kb   = (lane >> 4) << 3;
  const bf16_t* q = p + (size_t)(row0 + r) * ld + koff + kb;
  v8bf lo = *(const v8bf*)(q);
  v8bf hi = *(const v8bf*)(q + 16);
  v16bf f;
#pragma unroll
  for (int i = 0; i < 8; ++i) { f[i] = lo[i]; f[i + 8] = hi[i]; }
  return f;
}

// ---------------------------------------------------------------------------
// fp32 -> bf16 conversion
// ---------------------------------------------------------------------------
__global__ void cvt_f32_bf16(const float* __restrict__ in,
                             bf16_t* __restrict__ out, int n) {
  int i = (blockIdx.x * blockDim.x + threadIdx.x) * 4;
  if (i + 3 < n) {
    float4 f = *(const float4*)(in + i);
    out[i + 0] = (bf16_t)f.x;
    out[i + 1] = (bf16_t)f.y;
    out[i + 2] = (bf16_t)f.z;
    out[i + 3] = (bf16_t)f.w;
  } else {
    for (; i < n; ++i) out[i] = (bf16_t)in[i];
  }
}

// ---------------------------------------------------------------------------
// C[M,N] = A[M,K] @ B[N,K]^T + bias, bf16 WMMA, f32 accum.
// Block: 256 thr (8 waves) -> 128x256 C tile; wave tile 64x64 (4x4 frags).
// K-step 64 staged in LDS (As 16KB + Bs 32KB).
// ---------------------------------------------------------------------------
template <bool CLIP, bool OUTF32>
__global__ __launch_bounds__(256) void gemm_bf16_wmma(
    const bf16_t* __restrict__ A, const bf16_t* __restrict__ Bm,
    const float* __restrict__ bias, bf16_t* __restrict__ Cb,
    float* __restrict__ Cf, int Mdim, int Ndim, int Kdim) {
  constexpr int KT = 64;
  __shared__ bf16_t As[128 * KT];
  __shared__ bf16_t Bs[256 * KT];

  const int m0   = blockIdx.y * 128;
  const int n0   = blockIdx.x * 256;
  const int tid  = threadIdx.x;
  const int wave = tid >> 5;
  const int wm   = wave >> 2;   // 0..1 -> 64-row slice
  const int wn   = wave & 3;    // 0..3 -> 64-col slice

  v8f acc[4][4];
#pragma unroll
  for (int i = 0; i < 4; ++i)
#pragma unroll
    for (int j = 0; j < 4; ++j)
      acc[i][j] = (v8f){0.f,0.f,0.f,0.f,0.f,0.f,0.f,0.f};

  for (int k0 = 0; k0 < Kdim; k0 += KT) {
    // stage A: 128x64 = 1024 v8bf chunks, 4 per thread
#pragma unroll
    for (int it = 0; it < 4; ++it) {
      const int idx = tid + 256 * it;
      const int rr = idx >> 3, c8 = (idx & 7) * 8;
      const bf16_t* src = A + (size_t)(m0 + rr) * Kdim + k0 + c8;
      *(v8bf*)(As + rr * KT + c8) = *(const v8bf*)(src);
      if (it == 0 && k0 + KT < Kdim) __builtin_prefetch(src + KT, 0, 1);
    }
    // stage B: 256x64 = 2048 v8bf chunks, 8 per thread
#pragma unroll
    for (int it = 0; it < 8; ++it) {
      const int idx = tid + 256 * it;
      const int rr = idx >> 3, c8 = (idx & 7) * 8;
      const bf16_t* src = Bm + (size_t)(n0 + rr) * Kdim + k0 + c8;
      *(v8bf*)(Bs + rr * KT + c8) = *(const v8bf*)(src);
      if (it == 0 && k0 + KT < Kdim) __builtin_prefetch(src + KT, 0, 1);
    }
    __syncthreads();

#pragma unroll
    for (int kk = 0; kk < 2; ++kk) {       // two 32-deep WMMA steps
      v16bf a[4];
#pragma unroll
      for (int i = 0; i < 4; ++i)
        a[i] = frag_ld(As, wm * 64 + i * 16, KT, kk * 32);
#pragma unroll
      for (int j = 0; j < 4; ++j) {
        v16bf bf = frag_ld(Bs, wn * 64 + j * 16, KT, kk * 32);
#pragma unroll
        for (int i = 0; i < 4; ++i)
          acc[i][j] = wmma_bf16(a[i], bf, acc[i][j]);
      }
    }
    __syncthreads();
  }

  const int lane = tid & 31;
  const int nn   = lane & 15;
  const int mb   = (lane >> 4) << 3;
#pragma unroll
  for (int i = 0; i < 4; ++i) {
#pragma unroll
    for (int j = 0; j < 4; ++j) {
      const int col = n0 + wn * 64 + j * 16 + nn;
      const float bv = bias ? bias[col] : 0.0f;
#pragma unroll
      for (int v = 0; v < 8; ++v) {
        const int row = m0 + wm * 64 + i * 16 + mb + v;
        float val = acc[i][j][v] + bv;
        if (CLIP) val = fminf(fmaxf(val, -CLIPV), CLIPV);
        if (OUTF32) Cf[(size_t)row * Ndim + col] = val;
        else        Cb[(size_t)row * Ndim + col] = (bf16_t)val;
      }
    }
  }
}

// ---------------------------------------------------------------------------
// LayerNorm over D=2048 for the q or k slice of qkv [M x 3D], in place.
// One block per row; fp32 math, bf16 storage.
// ---------------------------------------------------------------------------
__global__ __launch_bounds__(256) void qk_layernorm(
    bf16_t* __restrict__ qkv, const float* __restrict__ g,
    const float* __restrict__ bb, int off) {
  __shared__ float red[256];
  const int tid = threadIdx.x;
  bf16_t* p = qkv + (size_t)blockIdx.x * (3 * DD) + off;

  float x[8];
  float s = 0.0f;
#pragma unroll
  for (int e = 0; e < 8; ++e) { x[e] = (float)p[e * 256 + tid]; s += x[e]; }
  red[tid] = s;
  __syncthreads();
  for (int o = 128; o > 0; o >>= 1) {
    if (tid < o) red[tid] += red[tid + o];
    __syncthreads();
  }
  const float mean = red[0] * (1.0f / DD);
  __syncthreads();

  float vs = 0.0f;
#pragma unroll
  for (int e = 0; e < 8; ++e) { float d = x[e] - mean; vs += d * d; }
  red[tid] = vs;
  __syncthreads();
  for (int o = 128; o > 0; o >>= 1) {
    if (tid < o) red[tid] += red[tid + o];
    __syncthreads();
  }
  const float rstd = rsqrtf(red[0] * (1.0f / DD) + 1e-5f);
#pragma unroll
  for (int e = 0; e < 8; ++e) {
    const int idx = e * 256 + tid;
    p[idx] = (bf16_t)((x[e] - mean) * rstd * g[idx] + bb[idx]);
  }
}

// ---------------------------------------------------------------------------
// Flash attention: grid (S/64, H, B), 128 threads (4 waves), each wave owns
// 16 q rows. KV streamed in 64-key tiles; online softmax; WMMA for QK^T and
// P@V (V staged transposed in LDS so the key dim is contraction-contiguous).
// ---------------------------------------------------------------------------
__global__ __launch_bounds__(128) void flash_attn_wmma(
    const bf16_t* __restrict__ qkv,        // [M x 3D], q|k|v, q/k LN'ed
    const float* __restrict__ attn_bias,   // [H*S]
    const unsigned char* __restrict__ kpm, // [B*S] bool
    bf16_t* __restrict__ ctx) {            // [M x D]
  constexpr int QT = 64, KT = 64;
  __shared__ bf16_t Qs[QT * DHH];
  __shared__ bf16_t Ks[KT * DHH];
  __shared__ bf16_t Vt[DHH * KT];        // transposed: [dh][key]
  __shared__ bf16_t Ps[4][16 * KT];      // per-wave P staging

  const int b = blockIdx.z, h = blockIdx.y;
  const int q0 = blockIdx.x * QT;
  const int tid = threadIdx.x, wave = tid >> 5, lane = tid & 31;
  const size_t rs = 3 * DD;              // qkv row stride
  const int nn = lane & 15;
  const int mb = (lane >> 4) << 3;
  const float scale = 0.088388347648318447f;  // 1/sqrt(128)

  // Load the Q tile once.
  for (int idx = tid; idx < QT * DHH / 8; idx += 128) {
    const int rr = idx >> 4, c8 = idx & 15;
    *(v8bf*)(Qs + rr * DHH + c8 * 8) =
        *(const v8bf*)(qkv + (size_t)(b * SS + q0 + rr) * rs + h * DHH + c8 * 8);
  }
  __syncthreads();

  v16bf aq[4];
#pragma unroll
  for (int kk = 0; kk < 4; ++kk) aq[kk] = frag_ld(Qs, wave * 16, DHH, kk * 32);

  float mstate[8], lstate[8];
  v8f acc[8];
#pragma unroll
  for (int v = 0; v < 8; ++v) { mstate[v] = NEGBIG; lstate[v] = 0.0f; }
#pragma unroll
  for (int d = 0; d < 8; ++d) acc[d] = (v8f){0.f,0.f,0.f,0.f,0.f,0.f,0.f,0.f};

  const int kend = q0 + QT;   // causal upper bound (exclusive), block-uniform
  for (int k0 = 0; k0 < kend; k0 += KT) {
    __syncthreads();  // previous iteration finished with Ks/Vt
    for (int idx = tid; idx < KT * DHH / 8; idx += 128) {
      const int rr = idx >> 4, c8 = idx & 15;
      *(v8bf*)(Ks + rr * DHH + c8 * 8) =
          *(const v8bf*)(qkv + (size_t)(b * SS + k0 + rr) * rs + DD + h * DHH + c8 * 8);
      v8bf vv =
          *(const v8bf*)(qkv + (size_t)(b * SS + k0 + rr) * rs + 2 * DD + h * DHH + c8 * 8);
#pragma unroll
      for (int e = 0; e < 8; ++e) Vt[(c8 * 8 + e) * KT + rr] = vv[e];
    }
    __syncthreads();

    // scores = Q @ K^T for 16 q-rows x 64 keys
    v8f sc[4];
#pragma unroll
    for (int j = 0; j < 4; ++j) {
      sc[j] = (v8f){0.f,0.f,0.f,0.f,0.f,0.f,0.f,0.f};
#pragma unroll
      for (int kk = 0; kk < 4; ++kk) {
        v16bf bk = frag_ld(Ks, j * 16, DHH, kk * 32);
        sc[j] = wmma_bf16(aq[kk], bk, sc[j]);
      }
    }

    // epilogue: scale, bias, padding + causal mask, row max
    float rowmax[8];
#pragma unroll
    for (int v = 0; v < 8; ++v) rowmax[v] = NEGBIG;
#pragma unroll
    for (int j = 0; j < 4; ++j) {
      const int key = k0 + j * 16 + nn;
      const float bv = attn_bias[h * SS + key];
      const bool pad_ok = kpm[b * SS + key] != 0;
#pragma unroll
      for (int v = 0; v < 8; ++v) {
        const int qrow = q0 + wave * 16 + mb + v;
        float val = sc[j][v] * scale + bv;
        if (!pad_ok || key > qrow) val = NEGBIG;
        sc[j][v] = val;
        rowmax[v] = fmaxf(rowmax[v], val);
      }
    }
#pragma unroll
    for (int v = 0; v < 8; ++v) {
      float r = rowmax[v];
      r = fmaxf(r, __shfl_xor(r, 1, 32));
      r = fmaxf(r, __shfl_xor(r, 2, 32));
      r = fmaxf(r, __shfl_xor(r, 4, 32));
      r = fmaxf(r, __shfl_xor(r, 8, 32));
      rowmax[v] = r;
    }

    float corr[8], rowsum[8];
#pragma unroll
    for (int v = 0; v < 8; ++v) {
      const float mnew = fmaxf(mstate[v], rowmax[v]);
      corr[v] = __expf(mstate[v] - mnew);
      mstate[v] = mnew;
      rowsum[v] = 0.0f;
    }
#pragma unroll
    for (int j = 0; j < 4; ++j)
#pragma unroll
      for (int v = 0; v < 8; ++v) {
        const float p = __expf(sc[j][v] - mstate[v]);
        sc[j][v] = p;
        rowsum[v] += p;
      }
#pragma unroll
    for (int v = 0; v < 8; ++v) {
      float r = rowsum[v];
      r += __shfl_xor(r, 1, 32);
      r += __shfl_xor(r, 2, 32);
      r += __shfl_xor(r, 4, 32);
      r += __shfl_xor(r, 8, 32);
      lstate[v] = lstate[v] * corr[v] + r;
    }
#pragma unroll
    for (int d = 0; d < 8; ++d)
#pragma unroll
      for (int v = 0; v < 8; ++v) acc[d][v] *= corr[v];

    // stage P (C layout -> LDS -> A layout)
#pragma unroll
    for (int j = 0; j < 4; ++j)
#pragma unroll
      for (int v = 0; v < 8; ++v)
        Ps[wave][(mb + v) * KT + j * 16 + nn] = (bf16_t)sc[j][v];
    __syncthreads();

    v16bf ap0 = frag_ld(&Ps[wave][0], 0, KT, 0);
    v16bf ap1 = frag_ld(&Ps[wave][0], 0, KT, 32);
#pragma unroll
    for (int d = 0; d < 8; ++d) {
      v16bf bv0 = frag_ld(Vt, d * 16, KT, 0);
      v16bf bv1 = frag_ld(Vt, d * 16, KT, 32);
      acc[d] = wmma_bf16(ap0, bv0, acc[d]);
      acc[d] = wmma_bf16(ap1, bv1, acc[d]);
    }
  }

  // normalize and write ctx (bf16)
#pragma unroll
  for (int d = 0; d < 8; ++d) {
#pragma unroll
    for (int v = 0; v < 8; ++v) {
      const float inv = 1.0f / lstate[v];
      const int row = b * SS + q0 + wave * 16 + mb + v;
      ctx[(size_t)row * DD + h * DHH + d * 16 + nn] = (bf16_t)(acc[d][v] * inv);
    }
  }
}

// ---------------------------------------------------------------------------
// host launcher
// ---------------------------------------------------------------------------
extern "C" void kernel_launch(void* const* d_in, const int* in_sizes, int n_in,
                              void* d_out, int out_size, void* d_ws,
                              size_t ws_size, hipStream_t stream) {
  const float* x         = (const float*)d_in[0];
  const float* attn_bias = (const float*)d_in[1];
  const unsigned char* kpm = (const unsigned char*)d_in[2];
  const float* Wqkv_w    = (const float*)d_in[3];
  const float* Wqkv_b    = (const float*)d_in[4];
  const float* q_ln_g    = (const float*)d_in[5];
  const float* q_ln_b    = (const float*)d_in[6];
  const float* k_ln_g    = (const float*)d_in[7];
  const float* k_ln_b    = (const float*)d_in[8];
  const float* out_w     = (const float*)d_in[9];
  const float* out_b     = (const float*)d_in[10];
  float* out = (float*)d_out;

  bf16_t* xb    = (bf16_t*)d_ws;                       // M*D
  bf16_t* wqkvb = xb + (size_t)MM * DD;                // 3D*D
  bf16_t* woutb = wqkvb + (size_t)3 * DD * DD;         // D*D
  bf16_t* qkvb  = woutb + (size_t)DD * DD;             // M*3D
  bf16_t* ctxb  = qkvb + (size_t)MM * 3 * DD;          // M*D

  // 1) fp32 -> bf16
  {
    int n = MM * DD;
    cvt_f32_bf16<<<(n / 4 + 255) / 256, 256, 0, stream>>>(x, xb, n);
    n = 3 * DD * DD;
    cvt_f32_bf16<<<(n / 4 + 255) / 256, 256, 0, stream>>>(Wqkv_w, wqkvb, n);
    n = DD * DD;
    cvt_f32_bf16<<<(n / 4 + 255) / 256, 256, 0, stream>>>(out_w, woutb, n);
  }

  // 2) qkv = clip(x @ Wqkv^T + b)   [M=4096, N=6144, K=2048]
  {
    dim3 grid(3 * DD / 256, MM / 128);
    gemm_bf16_wmma<true, false><<<grid, 256, 0, stream>>>(
        xb, wqkvb, Wqkv_b, qkvb, nullptr, MM, 3 * DD, DD);
  }

  // 3) LayerNorm q and k slices (in place)
  qk_layernorm<<<MM, 256, 0, stream>>>(qkvb, q_ln_g, q_ln_b, 0);
  qk_layernorm<<<MM, 256, 0, stream>>>(qkvb, k_ln_g, k_ln_b, DD);

  // 4) flash attention -> ctx (bf16)
  {
    dim3 grid(SS / 64, HH, BB);
    flash_attn_wmma<<<grid, 128, 0, stream>>>(qkvb, attn_bias, kpm, ctxb);
  }

  // 5) out = ctx @ out_w^T + out_b  (fp32 -> d_out)  [M=4096, N=2048, K=2048]
  {
    dim3 grid(DD / 256, MM / 128);
    gemm_bf16_wmma<false, true><<<grid, 256, 0, stream>>>(
        ctxb, woutb, out_b, nullptr, out, MM, DD, DD);
  }
}
